// TransformerBlock_57200374448393
// MI455X (gfx1250) — matrix operand order repaired and checked by reading, not compiled, not run on hardware
//
#include <hip/hip_runtime.h>
#include <hip/hip_bf16.h>

// ---------------------------------------------------------------------------
// PointTransformerConv fused for MI455X (gfx1250, wave32, WMMA bf16 + TDM).
//
//   K_pack x9 : f32 weights -> bf16 in WMMA B-fragment order (workspace)
//   K_zero    : m / z / agg = 0
//   K_node    : h=relu(x@W_in+b);  q=h@W_dst, k=h@W_src, v=h@W_lin  (bf16)
//   K_edge    : delta=MLP2(pos[d]-pos[s]); alpha=MLP2(q[d]-k[s]+delta);
//               store delta,alpha (bf16); m[d,c] = atomic max alpha (alpha>=0)
//   K_scatter : ea=exp(alpha-m[d]); z[d]+=ea; agg[d]+=ea*(v[s]+delta)
//   K_out     : out = relu((agg/(z+eps)) @ W_out + b_out)   (f32)
//
// Weights are staged into a rotating LDS buffer by the Tensor Data Mover
// (tensor_load_to_lds + s_wait_tensorcnt), then consumed as 2x ds_load_b128
// B fragments; A fragments are 2x ds_load_b128 from per-wave LDS tiles.
// ---------------------------------------------------------------------------

typedef __attribute__((ext_vector_type(16))) __bf16 v16bf;
typedef __attribute__((ext_vector_type(8)))  float  v8f;
typedef __attribute__((ext_vector_type(4)))  unsigned int v4u;
typedef __attribute__((ext_vector_type(8)))  int v8i_t;
typedef __attribute__((ext_vector_type(4)))  int v4i_t;

#define C_DIM   128
#define LDP     136                    // padded LDS tile pitch (bf16), 272B/row
#define TILES_B (8 * 16 * LDP * 2)     // 34816 B : 8 per-wave 16x128 tiles
#define WBUF_B  (C_DIM * C_DIM * 2)    // 32768 B : one packed weight matrix
#define EIDX_B  (8 * 32 * 8)           // 2048  B : per-wave src/dst indices

#if defined(__has_builtin)
#if __has_builtin(__builtin_amdgcn_tensor_load_to_lds) && \
    __has_builtin(__builtin_amdgcn_s_wait_tensorcnt)
#define HAVE_TDM 1
#endif
#endif
#ifndef HAVE_TDM
#define HAVE_TDM 0
#endif

union frag_u { v16bf v; uint4 u[2]; };
union bf4_u  { uint2 u; __bf16 h[4]; };

// ---- A fragment (16x32 bf16 MxK): two contiguous 16B runs per lane ---------
__device__ __forceinline__ v16bf load_a_frag(const __bf16* tile, int lane, int kt) {
  int m = lane & 15, half = lane >> 4;
  const __bf16* row = tile + m * LDP + kt * 32 + half * 8;
  frag_u f;
  f.u[0] = *(const uint4*)(row);        // a[0..7]  = row[k0 .. k0+7]
  f.u[1] = *(const uint4*)(row + 16);   // a[8..15] = row[k0+16 .. k0+23]
  return f.v;
}

// ---- B fragment from packed LDS weight image [kt][nt][lane][16] ------------
__device__ __forceinline__ v16bf load_b_frag(const __bf16* wbuf, int lane,
                                             int kt, int nt) {
  const uint4* p = (const uint4*)(wbuf + ((((kt << 3) + nt) << 5) + lane) * 16);
  frag_u f;
  f.u[0] = p[0];
  f.u[1] = p[1];
  return f.v;
}

// ---- stage one packed weight (32KB contiguous) global -> LDS ---------------
// Tensor Data Mover path: one wave issues a 1-D D# (ISA 8.3/8.4):
//   group0: count=1 | lds_addr | global_addr[56:0] | type=2
//   group1: data_size=3 (8B), tensor_dim0 = tile_dim0 = 4096 elements
__device__ __forceinline__ void stage_weight(const __bf16* __restrict__ Wp,
                                             __bf16* wbuf, unsigned lds_off,
                                             int tid) {
#if HAVE_TDM
  if ((tid >> 5) == 0) {   // wave 0 drives the DMA (EXEC ignored by TDM)
    unsigned long long ga = (unsigned long long)(const void*)Wp;
    v4u g0;
    g0[0] = 1u;                                            // count=1 (valid)
    g0[1] = lds_off;                                       // lds_addr (bytes)
    g0[2] = (unsigned)(ga & 0xFFFFFFFFu);                  // global_addr lo
    g0[3] = (unsigned)((ga >> 32) & 0x01FFFFFFu) | (2u << 30);  // hi + type=2
    v8i_t g1;
    g1[0] = (3 << 16);                 // data_size = 8B, no mask/pad/iterate
    g1[1] = (int)(4096u << 16);        // tensor_dim0[15:0] = 4096
    g1[2] = 0;                         // tensor_dim0 hi / tensor_dim1 lo
    g1[3] = (int)(4096u << 16);        // tile_dim0 = 4096
    g1[4] = 0;                         // tile_dim1 / tile_dim2 unused
    g1[5] = 4096;                      // tensor_dim0_stride
    g1[6] = 0;
    g1[7] = 0;
    v4i_t z4 = {0, 0, 0, 0};
#if __clang_major__ >= 23
    v8i_t z8 = {0, 0, 0, 0, 0, 0, 0, 0};
    __builtin_amdgcn_tensor_load_to_lds(g0, g1, z4, z4, z8, 0);
#else
    __builtin_amdgcn_tensor_load_to_lds(g0, g1, z4, z4, 0);
#endif
    __builtin_amdgcn_s_wait_tensorcnt(0);
  }
  (void)wbuf;
#else
  const uint4* src = (const uint4*)Wp;
  uint4*       dst = (uint4*)wbuf;
#pragma unroll
  for (int i = 0; i < 8; ++i) dst[tid + 256 * i] = src[tid + 256 * i];
  (void)lds_off;
#endif
}

// ---- one wave: out[16x128] = act(A @ W + bias) -> LDS tile -----------------
__device__ __forceinline__ void wave_gemm(const v16bf a[4], const __bf16* wbuf,
                                          const float* __restrict__ bias,
                                          __bf16* out, int lane, bool relu) {
  int n15 = lane & 15, half = lane >> 4;
#pragma unroll
  for (int nt = 0; nt < 8; ++nt) {
    int col = nt * 16 + n15;
    float bv = bias ? bias[col] : 0.0f;
    v8f acc;
#pragma unroll
    for (int r = 0; r < 8; ++r) acc[r] = bv;
    // preload all 4 B fragments so ds loads overlap the WMMA chain
    v16bf b0 = load_b_frag(wbuf, lane, 0, nt);
    v16bf b1 = load_b_frag(wbuf, lane, 1, nt);
    v16bf b2 = load_b_frag(wbuf, lane, 2, nt);
    v16bf b3 = load_b_frag(wbuf, lane, 3, nt);
    acc = __builtin_amdgcn_wmma_f32_16x16x32_bf16(false, a[0], false, b0, (short)0, acc, false, false);
    acc = __builtin_amdgcn_wmma_f32_16x16x32_bf16(false, a[1], false, b1, (short)0, acc, false, false);
    acc = __builtin_amdgcn_wmma_f32_16x16x32_bf16(false, a[2], false, b2, (short)0, acc, false, false);
    acc = __builtin_amdgcn_wmma_f32_16x16x32_bf16(false, a[3], false, b3, (short)0, acc, false, false);
#pragma unroll
    for (int r = 0; r < 8; ++r) {
      float x = relu ? fmaxf(acc[r], 0.0f) : acc[r];
      out[(r + 8 * half) * LDP + col] = (__bf16)x;   // C: row r+8*half, col
    }
  }
}

// ---- one wave: relu(A @ W + bias) stored f32 straight to global ------------
__device__ __forceinline__ void wave_gemm_store_f32(const v16bf a[4],
                                                    const __bf16* wbuf,
                                                    const float* __restrict__ bias,
                                                    float* __restrict__ out,
                                                    int row0, int n, int lane) {
  int n15 = lane & 15, half = lane >> 4;
#pragma unroll
  for (int nt = 0; nt < 8; ++nt) {
    int col = nt * 16 + n15;
    float bv = bias ? bias[col] : 0.0f;
    v8f acc;
#pragma unroll
    for (int r = 0; r < 8; ++r) acc[r] = bv;
    v16bf b0 = load_b_frag(wbuf, lane, 0, nt);
    v16bf b1 = load_b_frag(wbuf, lane, 1, nt);
    v16bf b2 = load_b_frag(wbuf, lane, 2, nt);
    v16bf b3 = load_b_frag(wbuf, lane, 3, nt);
    acc = __builtin_amdgcn_wmma_f32_16x16x32_bf16(false, a[0], false, b0, (short)0, acc, false, false);
    acc = __builtin_amdgcn_wmma_f32_16x16x32_bf16(false, a[1], false, b1, (short)0, acc, false, false);
    acc = __builtin_amdgcn_wmma_f32_16x16x32_bf16(false, a[2], false, b2, (short)0, acc, false, false);
    acc = __builtin_amdgcn_wmma_f32_16x16x32_bf16(false, a[3], false, b3, (short)0, acc, false, false);
#pragma unroll
    for (int r = 0; r < 8; ++r) {
      int row = row0 + r + 8 * half;
      float x = fmaxf(acc[r], 0.0f);
      if (row < n) out[(size_t)row * C_DIM + col] = x;
    }
  }
}

// ---- cooperative LDS tile (bf16) -> global bf16 [row][128] -----------------
__device__ __forceinline__ void store_tile_bf16(const __bf16* tile,
                                                __bf16* __restrict__ g,
                                                long long row0, int lane,
                                                long long nrows) {
  int c0 = lane * 4;  // 8B per lane per row
  for (int e = 0; e < 16; ++e) {
    long long r = row0 + e;
    if (r < nrows)
      *(uint2*)(g + r * C_DIM + c0) = *(const uint2*)(tile + e * LDP + c0);
  }
}

// ===========================================================================
// pack f32 weight W[k][n] -> bf16 fragment order [kt][nt][lane][j]
__global__ __launch_bounds__(256) void k_pack(const float* __restrict__ w,
                                              __bf16* __restrict__ o) {
  int idx = blockIdx.x * 256 + threadIdx.x;     // 0 .. 16383
  int j    = idx & 15;
  int lane = (idx >> 4) & 31;
  int frag = idx >> 9;                          // kt*8 + nt
  int kt = frag >> 3, nt = frag & 7;
  int half = lane >> 4;
  int k = kt * 32 + half * 8 + (j & 7) + ((j >> 3) << 4);  // table order: 8h+j | 16+8h+(j-8)
  int n = nt * 16 + (lane & 15);
  o[idx] = (__bf16)w[k * C_DIM + n];
}

__global__ __launch_bounds__(256) void k_zero(float* __restrict__ p, long long count) {
  long long i = (long long)blockIdx.x * 256 + threadIdx.x;
  if (i < count) p[i] = 0.0f;
}

// ---------------- node kernel: h, then q/k/v (bf16) -------------------------
__global__ __launch_bounds__(256) void k_node(
    const float* __restrict__ x, const __bf16* __restrict__ WinP,
    const float* __restrict__ b_in, const __bf16* __restrict__ WdstP,
    const __bf16* __restrict__ WsrcP, const __bf16* __restrict__ WlinP,
    __bf16* __restrict__ qb, __bf16* __restrict__ kb, __bf16* __restrict__ vb,
    int n) {
  extern __shared__ char smem_dyn[];
  int tid = threadIdx.x, wave = tid >> 5, lane = tid & 31;
  __bf16* tile = (__bf16*)smem_dyn + wave * 16 * LDP;
  __bf16* wbuf = (__bf16*)(smem_dyn + TILES_B);
  int row0 = blockIdx.x * 128 + wave * 16;
  int c0 = lane * 4;

  __builtin_prefetch(WdstP, 0, 3);   // warm next weights while W_in stages
  __builtin_prefetch(WsrcP, 0, 3);
  __builtin_prefetch(WlinP, 0, 3);

  stage_weight(WinP, wbuf, TILES_B, tid);
  for (int e = 0; e < 16; ++e) {                 // stage x -> bf16 tile
    int r = row0 + e;
    float4 xv = (r < n) ? *(const float4*)(x + (size_t)r * C_DIM + c0)
                        : make_float4(0.f, 0.f, 0.f, 0.f);
    __bf16* t = tile + e * LDP + c0;
    t[0] = (__bf16)xv.x; t[1] = (__bf16)xv.y;
    t[2] = (__bf16)xv.z; t[3] = (__bf16)xv.w;
  }
  __syncthreads();
  v16bf a[4];
#pragma unroll
  for (int kt = 0; kt < 4; ++kt) a[kt] = load_a_frag(tile, lane, kt);
  __syncthreads();
  wave_gemm(a, wbuf, b_in, tile, lane, true);      // h
  __syncthreads();
  v16bf ah[4];
#pragma unroll
  for (int kt = 0; kt < 4; ++kt) ah[kt] = load_a_frag(tile, lane, kt);
  stage_weight(WdstP, wbuf, TILES_B, tid);
  __syncthreads();
  wave_gemm(ah, wbuf, nullptr, tile, lane, false);  // q
  __syncthreads();
  store_tile_bf16(tile, qb, row0, lane, n);
  stage_weight(WsrcP, wbuf, TILES_B, tid);
  __syncthreads();
  wave_gemm(ah, wbuf, nullptr, tile, lane, false);  // k
  __syncthreads();
  store_tile_bf16(tile, kb, row0, lane, n);
  stage_weight(WlinP, wbuf, TILES_B, tid);
  __syncthreads();
  wave_gemm(ah, wbuf, nullptr, tile, lane, false);  // v
  __syncthreads();
  store_tile_bf16(tile, vb, row0, lane, n);
}

// ---------------- edge kernel: delta, alpha, segment-max --------------------
__global__ __launch_bounds__(256) void k_edge(
    const long long* __restrict__ ei, const float* __restrict__ pos,
    const __bf16* __restrict__ qb, const __bf16* __restrict__ kb,
    const __bf16* __restrict__ pw1P, const float* __restrict__ pb1,
    const __bf16* __restrict__ pw2P, const float* __restrict__ pb2,
    const __bf16* __restrict__ aw1P, const float* __restrict__ ab1,
    const __bf16* __restrict__ aw2P, const float* __restrict__ ab2,
    __bf16* __restrict__ delta_g, __bf16* __restrict__ alpha_g,
    unsigned int* __restrict__ m, long long E) {
  extern __shared__ char smem_dyn[];
  int tid = threadIdx.x, wave = tid >> 5, lane = tid & 31;
  __bf16*    tile = (__bf16*)smem_dyn + wave * 16 * LDP;
  __bf16*    wbuf = (__bf16*)(smem_dyn + TILES_B);
  long long* eidx = (long long*)(smem_dyn + TILES_B + WBUF_B) + wave * 32;
  long long e0 = (long long)blockIdx.x * 128 + wave * 16;
  int c0 = lane * 4;

  __builtin_prefetch(pw2P, 0, 3);
  __builtin_prefetch(aw1P, 0, 3);
  __builtin_prefetch(aw2P, 0, 3);

  stage_weight(pw1P, wbuf, TILES_B, tid);
  {  // per-wave edge indices: eidx[0..15]=src, eidx[16..31]=dst
    long long ee = e0 + (lane & 15);
    long long vsrc = (ee < E) ? ((lane < 16) ? ei[ee] : ei[E + ee]) : 0;
    eidx[lane] = vsrc;
  }
  __syncthreads();

  // stage pos[dst]-pos[src] -> bf16 tile
  for (int e = 0; e < 16; ++e) {
    __bf16* t = tile + e * LDP + c0;
    long long s = eidx[e], d = eidx[16 + e];
    float4 pd = *(const float4*)(pos + d * C_DIM + c0);
    float4 ps = *(const float4*)(pos + s * C_DIM + c0);
    t[0] = (__bf16)(pd.x - ps.x); t[1] = (__bf16)(pd.y - ps.y);
    t[2] = (__bf16)(pd.z - ps.z); t[3] = (__bf16)(pd.w - ps.w);
  }
  __syncthreads();
  v16bf a[4];
#pragma unroll
  for (int kt = 0; kt < 4; ++kt) a[kt] = load_a_frag(tile, lane, kt);
  __syncthreads();
  wave_gemm(a, wbuf, pb1, tile, lane, true);
  __syncthreads();
#pragma unroll
  for (int kt = 0; kt < 4; ++kt) a[kt] = load_a_frag(tile, lane, kt);
  stage_weight(pw2P, wbuf, TILES_B, tid);
  __syncthreads();
  wave_gemm(a, wbuf, pb2, tile, lane, true);        // delta (>=0) in tile
  __syncthreads();

  // store delta, build attn input in place: q[d]-k[s]+delta
  for (int e = 0; e < 16; ++e) {
    if (e0 + e >= E) continue;
    long long s = eidx[e], d = eidx[16 + e];
    __bf16* t = tile + e * LDP + c0;
    *(uint2*)(delta_g + (e0 + e) * C_DIM + c0) = *(const uint2*)t;
    bf4_u qr, kr;
    qr.u = *(const uint2*)(qb + d * C_DIM + c0);
    kr.u = *(const uint2*)(kb + s * C_DIM + c0);
#pragma unroll
    for (int j = 0; j < 4; ++j)
      t[j] = (__bf16)((float)qr.h[j] - (float)kr.h[j] + (float)t[j]);
  }
  stage_weight(aw1P, wbuf, TILES_B, tid);
  __syncthreads();
#pragma unroll
  for (int kt = 0; kt < 4; ++kt) a[kt] = load_a_frag(tile, lane, kt);
  __syncthreads();
  wave_gemm(a, wbuf, ab1, tile, lane, true);
  __syncthreads();
#pragma unroll
  for (int kt = 0; kt < 4; ++kt) a[kt] = load_a_frag(tile, lane, kt);
  stage_weight(aw2P, wbuf, TILES_B, tid);
  __syncthreads();
  wave_gemm(a, wbuf, ab2, tile, lane, true);        // alpha (>=0) in tile
  __syncthreads();

  // store alpha + per-(dst,channel) max via integer-bit atomicMax (alpha>=0)
  for (int e = 0; e < 16; ++e) {
    if (e0 + e >= E) continue;
    long long d = eidx[16 + e];
    const __bf16* t = tile + e * LDP + c0;
    *(uint2*)(alpha_g + (e0 + e) * C_DIM + c0) = *(const uint2*)t;
#pragma unroll
    for (int j = 0; j < 4; ++j)
      atomicMax(&m[d * C_DIM + c0 + j], __float_as_uint((float)t[j]));
  }
}

// ---------------- scatter kernel: ea, z, unnormalized agg (4 ch/thread) -----
__global__ __launch_bounds__(256) void k_scatter(
    const long long* __restrict__ ei, const __bf16* __restrict__ alpha_g,
    const __bf16* __restrict__ delta_g, const __bf16* __restrict__ vb,
    const float* __restrict__ m, float* __restrict__ z,
    float* __restrict__ agg, long long E) {
  long long gid = (long long)blockIdx.x * 256 + threadIdx.x;
  if (gid >= E * 32) return;
  long long e = gid >> 5;
  int c0 = (int)(gid & 31) * 4;
  long long s = ei[e], d = ei[E + e];
  long long eb = e * C_DIM + c0;
  long long db = d * C_DIM + c0;
  bf4_u av, dv, vv;
  av.u = *(const uint2*)(alpha_g + eb);
  dv.u = *(const uint2*)(delta_g + eb);
  vv.u = *(const uint2*)(vb + s * C_DIM + c0);
  float4 mv = *(const float4*)(m + db);
  const float mm[4] = {mv.x, mv.y, mv.z, mv.w};
#pragma unroll
  for (int j = 0; j < 4; ++j) {
    float ea = __expf((float)av.h[j] - mm[j]);
    unsafeAtomicAdd(&z[db + j], ea);
    unsafeAtomicAdd(&agg[db + j], ea * ((float)vv.h[j] + (float)dv.h[j]));
  }
}

// ---------------- output kernel: relu((agg/(z+eps)) @ W_out + b_out) --------
__global__ __launch_bounds__(256) void k_out(
    const float* __restrict__ agg, const float* __restrict__ z,
    const __bf16* __restrict__ WoutP, const float* __restrict__ b_out,
    float* __restrict__ out, int n) {
  extern __shared__ char smem_dyn[];
  int tid = threadIdx.x, wave = tid >> 5, lane = tid & 31;
  __bf16* tile = (__bf16*)smem_dyn + wave * 16 * LDP;
  __bf16* wbuf = (__bf16*)(smem_dyn + TILES_B);
  int row0 = blockIdx.x * 128 + wave * 16;
  int c0 = lane * 4;

  stage_weight(WoutP, wbuf, TILES_B, tid);
  for (int e = 0; e < 16; ++e) {
    int r = row0 + e;
    __bf16* t = tile + e * LDP + c0;
    if (r < n) {
      float4 av = *(const float4*)(agg + (size_t)r * C_DIM + c0);
      float4 zv = *(const float4*)(z + (size_t)r * C_DIM + c0);
      t[0] = (__bf16)(av.x / (zv.x + 1e-16f));
      t[1] = (__bf16)(av.y / (zv.y + 1e-16f));
      t[2] = (__bf16)(av.z / (zv.z + 1e-16f));
      t[3] = (__bf16)(av.w / (zv.w + 1e-16f));
    } else {
      t[0] = (__bf16)0.f; t[1] = (__bf16)0.f; t[2] = (__bf16)0.f; t[3] = (__bf16)0.f;
    }
  }
  __syncthreads();
  v16bf a[4];
#pragma unroll
  for (int kt = 0; kt < 4; ++kt) a[kt] = load_a_frag(tile, lane, kt);
  __syncthreads();
  wave_gemm_store_f32(a, wbuf, b_out, out, row0, n, lane);
}

// ===========================================================================
extern "C" void kernel_launch(void* const* d_in, const int* in_sizes, int n_in,
                              void* d_out, int out_size, void* d_ws, size_t ws_size,
                              hipStream_t stream) {
  const float*     x    = (const float*)d_in[0];
  const float*     pos  = (const float*)d_in[1];
  const long long* ei   = (const long long*)d_in[2];   // int64 edge_index [2,E]
  const float*     W_in = (const float*)d_in[3];
  const float*     b_in = (const float*)d_in[4];
  const float*     W_lin= (const float*)d_in[5];
  const float*     W_src= (const float*)d_in[6];
  const float*     W_dst= (const float*)d_in[7];
  const float*     pw1  = (const float*)d_in[8];
  const float*     pb1  = (const float*)d_in[9];
  const float*     pw2  = (const float*)d_in[10];
  const float*     pb2  = (const float*)d_in[11];
  const float*     aw1  = (const float*)d_in[12];
  const float*     ab1  = (const float*)d_in[13];
  const float*     aw2  = (const float*)d_in[14];
  const float*     ab2  = (const float*)d_in[15];
  const float*     W_out= (const float*)d_in[16];
  const float*     b_out= (const float*)d_in[17];
  float* out = (float*)d_out;

  const int       N = in_sizes[0] / C_DIM;
  const long long E = (long long)in_sizes[2] / 2;
  const int       WSZ = C_DIM * C_DIM;  // 16384

  // ---- workspace carve-out (256B aligned) ----
  char* p = (char*)d_ws;
  auto carve = [&p](size_t bytes) -> void* {
    void* r = (void*)p;
    p += (bytes + 255) & ~(size_t)255;
    return r;
  };
  __bf16* WinP  = (__bf16*)carve(WSZ * 2);
  __bf16* WdstP = (__bf16*)carve(WSZ * 2);
  __bf16* WsrcP = (__bf16*)carve(WSZ * 2);
  __bf16* WlinP = (__bf16*)carve(WSZ * 2);
  __bf16* pw1P  = (__bf16*)carve(WSZ * 2);
  __bf16* pw2P  = (__bf16*)carve(WSZ * 2);
  __bf16* aw1P  = (__bf16*)carve(WSZ * 2);
  __bf16* aw2P  = (__bf16*)carve(WSZ * 2);
  __bf16* WoutP = (__bf16*)carve(WSZ * 2);
  __bf16* qb    = (__bf16*)carve((size_t)N * C_DIM * 2);
  __bf16* kb    = (__bf16*)carve((size_t)N * C_DIM * 2);
  __bf16* vb    = (__bf16*)carve((size_t)N * C_DIM * 2);
  __bf16* deltb = (__bf16*)carve((size_t)E * C_DIM * 2);
  __bf16* alphb = (__bf16*)carve((size_t)E * C_DIM * 2);
  float*  mzagg = (float*)carve((size_t)3 * N * C_DIM * 4);  // m | z | agg
  float*  m   = mzagg;
  float*  zb  = mzagg + (size_t)N * C_DIM;
  float*  agg = mzagg + (size_t)2 * N * C_DIM;

  // ---- pack weights into fragment order (f32 -> bf16) ----
  const int pkB = WSZ / 256;
  k_pack<<<pkB, 256, 0, stream>>>(W_in,  WinP);
  k_pack<<<pkB, 256, 0, stream>>>(W_dst, WdstP);
  k_pack<<<pkB, 256, 0, stream>>>(W_src, WsrcP);
  k_pack<<<pkB, 256, 0, stream>>>(W_lin, WlinP);
  k_pack<<<pkB, 256, 0, stream>>>(pw1,   pw1P);
  k_pack<<<pkB, 256, 0, stream>>>(pw2,   pw2P);
  k_pack<<<pkB, 256, 0, stream>>>(aw1,   aw1P);
  k_pack<<<pkB, 256, 0, stream>>>(aw2,   aw2P);
  k_pack<<<pkB, 256, 0, stream>>>(W_out, WoutP);

  // ---- zero m/z/agg (alpha>=0 so m=0 matches segment_max semantics) ----
  long long zc = (long long)3 * N * C_DIM;
  k_zero<<<(unsigned)((zc + 255) / 256), 256, 0, stream>>>(mzagg, zc);

  // ---- node features ----
  k_node<<<(N + 127) / 128, 256, TILES_B + WBUF_B, stream>>>(
      x, WinP, b_in, WdstP, WsrcP, WlinP, qb, kb, vb, N);
  // ---- edge pass A: delta, alpha, segment max ----
  k_edge<<<(unsigned)((E + 127) / 128), 256, TILES_B + WBUF_B + EIDX_B, stream>>>(
      ei, pos, qb, kb, pw1P, pb1, pw2P, pb2, aw1P, ab1, aw2P, ab2,
      deltb, alphb, (unsigned int*)m, E);
  // ---- edge pass B: exp + scatter (deferred normalization) ----
  long long sc = E * 32;
  k_scatter<<<(unsigned)((sc + 255) / 256), 256, 0, stream>>>(
      ei, alphb, deltb, vb, m, zb, agg, E);
  // ---- output GEMM ----
  k_out<<<(N + 127) / 128, 256, TILES_B + WBUF_B, stream>>>(
      agg, zb, WoutP, b_out, out, N);
}
